// Concat_CrossAttnBlock_LearnedMask_51848845197803
// MI455X (gfx1250) — compile-verified, matrix operand order checked by reading, not run on hardware
//
#include <hip/hip_runtime.h>
#include <hip/hip_bf16.h>
#include <math.h>

#define DEV static __device__ __forceinline__

typedef __attribute__((ext_vector_type(16))) __bf16         v16bf;
typedef __attribute__((ext_vector_type(8)))  float          v8f;
typedef __attribute__((ext_vector_type(4)))  float          v4f;
typedef __attribute__((ext_vector_type(8)))  unsigned short us8;

// problem sizes
#define B_  2
#define C_  128
#define H_  256
#define W_  512
#define NC_ 512
#define DC_ 512
#define DA_ 256

// ---------- bf16 helpers (bit-level, round-to-nearest-even) ----------
DEV unsigned short f2b(float f){
  unsigned u = __builtin_bit_cast(unsigned, f);
  unsigned r = u + 0x7FFFu + ((u >> 16) & 1u);
  return (unsigned short)(r >> 16);
}
DEV __bf16 b2h(unsigned short s){ return __builtin_bit_cast(__bf16, s); }
DEV float  b2f(unsigned short s){ return __builtin_bit_cast(float, ((unsigned)s) << 16); }

// Build a 16x16x32 bf16 A/B fragment from a K-contiguous row pointer.
// Lane layout: element i -> K = (i>>3)*16 + hl*8 + (i&7): two contiguous
// 8-element (16B) runs at p+hl*8 and p+16+hl*8 -> two b128 loads.
DEV v16bf load_frag(const unsigned short* p, int hl){
  us8 lo = *(const us8*)(p + hl*8);
  us8 hi = *(const us8*)(p + 16 + hl*8);
  v16bf r;
#pragma unroll
  for (int i = 0; i < 8; ++i){ r[i] = b2h(lo[i]); r[8+i] = b2h(hi[i]); }
  return r;
}
DEV v16bf zero_frag(){
  v16bf r;
#pragma unroll
  for (int i = 0; i < 16; ++i) r[i] = b2h((unsigned short)0);
  return r;
}
DEV v8f wmma_bf16(v16bf a, v16bf b, v8f c){
  return __builtin_amdgcn_wmma_f32_16x16x32_bf16(false, a, false, b, (short)0, c, false, false);
}

// ---------------------------------------------------------------------
// prep: zero GN accumulators
// ---------------------------------------------------------------------
__global__ void zero_stats_kernel(float* __restrict__ gsum, float* __restrict__ gsumsq){
  int i = threadIdx.x; gsum[i] = 0.f; gsumsq[i] = 0.f;
}

// ---------------------------------------------------------------------
// prep: f32 -> bf16 straight convert (v matrix)
// ---------------------------------------------------------------------
__global__ void cvt_kernel(const float* __restrict__ src, unsigned short* __restrict__ dst, int n){
  for (int i = blockIdx.x*blockDim.x + threadIdx.x; i < n; i += gridDim.x*blockDim.x)
    dst[i] = f2b(src[i]);
}

// ---------------------------------------------------------------------
// prep: transpose+convert weight: dst[n*K + k] = bf16(src[k*N + n])
// ---------------------------------------------------------------------
__global__ void transpose_cvt_kernel(const float* __restrict__ src, unsigned short* __restrict__ dst,
                                     int K, int N){
  int idx = blockIdx.x*blockDim.x + threadIdx.x;
  if (idx < K*N){
    int n = idx / K, k = idx - n*K;
    dst[idx] = f2b(src[(size_t)k*N + n]);
  }
}

// ---------------------------------------------------------------------
// prep: conv weights [oc,ic,3,3] -> cwT[(tap*128+oc)*256+ic] bf16
// ---------------------------------------------------------------------
__global__ void convw_prep_kernel(const float* __restrict__ conv_w, unsigned short* __restrict__ cwT){
  int idx = blockIdx.x*blockDim.x + threadIdx.x;  // 9*128*256
  if (idx < 9*C_*2*C_){
    int tap = idx / (C_*2*C_);
    int rem = idx - tap*(C_*2*C_);
    int oc  = rem / (2*C_);
    int ic  = rem - oc*(2*C_);
    cwT[idx] = f2b(conv_w[((size_t)oc*(2*C_) + ic)*9 + tap]);
  }
}

// ---------------------------------------------------------------------
// 1) NonAdaptiveLearnedMask -> binary[h]
// ---------------------------------------------------------------------
__global__ void mask_kernel(const float* __restrict__ mask_w, float* __restrict__ binary){
  __shared__ float s[H_];
  int i = threadIdx.x;
  float sg = 1.f / (1.f + __expf(-mask_w[i]));
  s[i] = sg;
  __syncthreads();
  int cnt = 0;
  for (int j = 0; j < H_; ++j) cnt += (s[j] > sg) ? 1 : 0;
  // k_keep = int(256*0.8) = 204 ; x >= kth-largest <=> rank < 204
  float wta  = (cnt < 204) ? sg : 0.f;
  float prob = 1.f / (1.f + __expf(-wta));
  binary[i] = (prob > 0.5f) ? 1.f : 0.f;
}

// binary_mask output [b,1,w,h]: v4f stores (h is the fast axis)
__global__ void mask_out_kernel(const float* __restrict__ binary, float* __restrict__ outm){
  int i4 = blockIdx.x * blockDim.x + threadIdx.x;   // over 65536 float4s
  int hb = (i4 * 4) & (H_ - 1);
  v4f t;
#pragma unroll
  for (int j = 0; j < 4; ++j) t[j] = binary[hb + j];
  *(v4f*)&outm[(size_t)i4 * 4] = t;
}

// ---------------------------------------------------------------------
// 2) K/V projections (WMMA): Kp[b*512+n][d] ; VpT[(b*256+d)][n]  (bf16)
//    grid (64, 16, 2), block 32
// ---------------------------------------------------------------------
__global__ void __launch_bounds__(32) kv_proj_kernel(
    const unsigned short* __restrict__ vb, const unsigned short* __restrict__ wkT,
    const unsigned short* __restrict__ wvT, unsigned short* __restrict__ Kp,
    unsigned short* __restrict__ VpT)
{
  int l = threadIdx.x; int nl = l & 15, hl = l >> 4;
  const unsigned short* wT = blockIdx.z ? wvT : wkT;
  int row = blockIdx.x * 16 + nl;       // b*n_cond
  int col = blockIdx.y * 16 + nl;       // d_attn
  v8f acc;
#pragma unroll
  for (int r = 0; r < 8; ++r) acc[r] = 0.f;
  for (int kc = 0; kc < DC_/32; ++kc){
    v16bf a = load_frag(&vb[(size_t)row*DC_ + kc*32], hl);
    v16bf b = load_frag(&wT[(size_t)col*DC_ + kc*32], hl);
    acc = wmma_bf16(a, b, acc);
  }
  if (blockIdx.z == 0){
#pragma unroll
    for (int r = 0; r < 8; ++r)
      Kp[(size_t)(blockIdx.x*16 + r + 8*hl)*DA_ + col] = f2b(acc[r]);
  } else {
#pragma unroll
    for (int r = 0; r < 8; ++r){
      int orow = blockIdx.x*16 + r + 8*hl;          // b*512 + n
      int bb = orow >> 9, nn = orow & (NC_-1);
      VpT[(size_t)(bb*DA_ + col)*NC_ + nn] = f2b(acc[r]);
    }
  }
}

// ---------------------------------------------------------------------
// 3) Fused flash attention per (b,c,16 w-rows)  grid (32,128,2), block 32
// ---------------------------------------------------------------------
__global__ void __launch_bounds__(32) attn_kernel(
    const float* __restrict__ x, const unsigned short* __restrict__ Kp,
    const unsigned short* __restrict__ VpT, const unsigned short* __restrict__ wqT,
    const unsigned short* __restrict__ woT, const float* __restrict__ bo,
    unsigned short* __restrict__ attnf)
{
  __shared__ __align__(16) unsigned short xt[16*DA_];
  __shared__ __align__(16) unsigned short qt[16*DA_];
  __shared__ __align__(16) unsigned short ot[16*DA_];
  __shared__ __align__(16) unsigned short pt[16*32];
  __shared__ __align__(16) unsigned short yt[16*16];   // Y store-staging tile
  __shared__ float rmax[16], rsum[16], nmax_s[16], scl_s[16], inv_s[16];

  int l = threadIdx.x; int nl = l & 15, hl = l >> 4;
  int w0 = blockIdx.x * 16;
  int c  = blockIdx.y;
  int b  = blockIdx.z;
  const float* xbc = x + (size_t)(b*C_ + c) * H_ * W_;
  __builtin_prefetch(wqT, 0, 2);
  __builtin_prefetch(woT, 0, 2);

  // stage X^T tile (16 w-rows x 256 h-cols) in LDS as bf16
  for (int idx = l; idx < 16*DA_; idx += 32){
    int hh = idx >> 4, m = idx & 15;
    xt[m*DA_ + hh] = f2b(xbc[(size_t)hh*W_ + w0 + m]);
  }
  if (l < 16){ rmax[l] = -INFINITY; rsum[l] = 0.f; }
  __syncthreads();

  // ---- Q = xt @ Wq -> qt ----
  {
    v16bf axf[8];
#pragma unroll
    for (int kc = 0; kc < 8; ++kc) axf[kc] = load_frag(&xt[nl*DA_ + kc*32], hl);
    for (int nd = 0; nd < DA_/16; ++nd){
      v8f acc;
#pragma unroll
      for (int r = 0; r < 8; ++r) acc[r] = 0.f;
#pragma unroll
      for (int kc = 0; kc < 8; ++kc){
        v16bf bfr = load_frag(&wqT[(size_t)(nd*16 + nl)*DA_ + kc*32], hl);
        acc = wmma_bf16(axf[kc], bfr, acc);
      }
#pragma unroll
      for (int r = 0; r < 8; ++r)
        qt[(r + 8*hl)*DA_ + nd*16 + nl] = f2b(acc[r]);
    }
  }
  __syncthreads();

  v8f accO[16];
#pragma unroll
  for (int nd = 0; nd < 16; ++nd)
#pragma unroll
    for (int r = 0; r < 8; ++r) accO[nd][r] = 0.f;

  // Q fragments are invariant over the flash loop: load once, reuse 16x
  v16bf aqf[8];
#pragma unroll
  for (int kc = 0; kc < 8; ++kc) aqf[kc] = load_frag(&qt[nl*DA_ + kc*32], hl);

  const unsigned short* Kb  = Kp  + (size_t)b * NC_ * DA_;
  const unsigned short* Vbt = VpT + (size_t)b * DA_ * NC_;

  for (int nc = 0; nc < NC_; nc += 32){
    // S chunk: two 16x16 tiles (no 1/sqrt(d): ref omits the scale)
    v8f sa, sb;
#pragma unroll
    for (int r = 0; r < 8; ++r){ sa[r] = 0.f; sb[r] = 0.f; }
#pragma unroll
    for (int kc = 0; kc < 8; ++kc){
      v16bf bk0 = load_frag(&Kb[(size_t)(nc + nl)*DA_ + kc*32], hl);
      v16bf bk1 = load_frag(&Kb[(size_t)(nc + 16 + nl)*DA_ + kc*32], hl);
      sa = wmma_bf16(aqf[kc], bk0, sa);
      sb = wmma_bf16(aqf[kc], bk1, sb);
    }
    // chunk row-max (D-layout row r+8*hl lives in one 16-lane half)
#pragma unroll
    for (int r = 0; r < 8; ++r){
      float vmx = fmaxf(sa[r], sb[r]);
      for (int off = 8; off >= 1; off >>= 1)
        vmx = fmaxf(vmx, __shfl_xor(vmx, off, 16));
      if (nl == 0){
        int row = r + 8*hl;
        float nm = fmaxf(rmax[row], vmx);
        nmax_s[row] = nm;
        scl_s[row]  = __expf(rmax[row] - nm);
        rmax[row]   = nm;
      }
    }
    __syncthreads();
    // P = exp(S - newmax), chunk row-sum, stage P bf16
#pragma unroll
    for (int r = 0; r < 8; ++r){
      int row = r + 8*hl;
      float nm = nmax_s[row];
      float pa = __expf(sa[r] - nm);
      float pb = __expf(sb[r] - nm);
      pt[row*32 + nl]      = f2b(pa);
      pt[row*32 + 16 + nl] = f2b(pb);
      float cs = pa + pb;
      for (int off = 8; off >= 1; off >>= 1)
        cs += __shfl_xor(cs, off, 16);
      if (nl == 0) rsum[row] = rsum[row] * scl_s[row] + cs;
    }
    // rescale running O
    float sc[8];
#pragma unroll
    for (int r = 0; r < 8; ++r) sc[r] = scl_s[r + 8*hl];
#pragma unroll
    for (int nd = 0; nd < 16; ++nd)
#pragma unroll
      for (int r = 0; r < 8; ++r) accO[nd][r] *= sc[r];
    __syncthreads();
    // O += P @ V_chunk (V transposed: K-contiguous)
    v16bf ap = load_frag(&pt[nl*32], hl);
#pragma unroll
    for (int nd = 0; nd < 16; ++nd){
      v16bf bv = load_frag(&Vbt[(size_t)(nd*16 + nl)*NC_ + nc], hl);
      accO[nd] = wmma_bf16(ap, bv, accO[nd]);
    }
    __syncthreads();
  }

  // normalize O and stage bf16 for output projection
  if (l < 16) inv_s[l] = 1.f / rsum[l];
  __syncthreads();
#pragma unroll
  for (int nd = 0; nd < 16; ++nd)
#pragma unroll
    for (int r = 0; r < 8; ++r){
      int row = r + 8*hl;
      ot[row*DA_ + nd*16 + nl] = f2b(accO[nd][r] * inv_s[row]);
    }
  __syncthreads();

  // Y = O @ Wo + bo -> attn_feature[b,c,h,w] (bf16), b128 stores via LDS tile
  unsigned short* af = attnf + (size_t)(b*C_ + c) * H_ * W_;
  v16bf aof[8];
#pragma unroll
  for (int kc = 0; kc < 8; ++kc) aof[kc] = load_frag(&ot[nl*DA_ + kc*32], hl);
  for (int nd = 0; nd < H_/16; ++nd){
    float bias = bo[nd*16 + nl];
    v8f acc;
#pragma unroll
    for (int r = 0; r < 8; ++r) acc[r] = bias;
#pragma unroll
    for (int kc = 0; kc < 8; ++kc){
      v16bf bw = load_frag(&woT[(size_t)(nd*16 + nl)*DA_ + kc*32], hl);
      acc = wmma_bf16(aof[kc], bw, acc);
    }
    // stage 16x16 Y tile in LDS [hcol][w], then one b128 store per half-lane row
#pragma unroll
    for (int r = 0; r < 8; ++r)
      yt[nl*16 + r + 8*hl] = f2b(acc[r]);
    __syncthreads();
    {
      int hcol = nd*16 + nl;
      us8 row8 = *(const us8*)&yt[nl*16 + 8*hl];
      *(us8*)&af[(size_t)hcol*W_ + w0 + 8*hl] = row8;
    }
    __syncthreads();
  }
}

// ---------------------------------------------------------------------
// 4a) NHWC repack: catN[b][h][w][ic] = bf16(cat([binary*x, attnf]))
//     LDS transpose for coalesced loads and b128 stores. grid (8,256,2), blk 256
// ---------------------------------------------------------------------
__global__ void repack_kernel(const float* __restrict__ x, const unsigned short* __restrict__ attnf,
                              const float* __restrict__ binary, unsigned short* __restrict__ catN)
{
  __shared__ __align__(16) unsigned short tile[64*(2*C_)];  // 32 KB
  int b = blockIdx.z, h = blockIdx.y, w0 = blockIdx.x * 64;
  float bsc = binary[h];
  // phase 1: coalesced channel-major reads -> LDS [w][ic]
  for (int idx = threadIdx.x; idx < 64*(2*C_); idx += blockDim.x){
    int ic = idx >> 6, wl = idx & 63;
    float v;
    if (ic < C_) v = bsc * x[((size_t)(b*C_ + ic)*H_ + h)*W_ + w0 + wl];
    else         v = b2f(attnf[((size_t)(b*C_ + ic - C_)*H_ + h)*W_ + w0 + wl]);
    tile[wl*(2*C_) + ic] = f2b(v);
  }
  __syncthreads();
  // phase 2: b128 stores of channel-contiguous rows
  unsigned short* orow = catN + ((size_t)(b*H_ + h)*W_ + w0) * (2*C_);
  for (int idx = threadIdx.x; idx < 64*(2*C_/8); idx += blockDim.x){
    int wl = idx >> 5, c8 = idx & 31;
    *(us8*)&orow[(size_t)wl*(2*C_) + c8*8] = *(const us8*)&tile[wl*(2*C_) + c8*8];
  }
}

// ---------------------------------------------------------------------
// 4b) conv3x3 implicit GEMM (WMMA) + fused GN sum/sumsq. grid(32,256,16), blk 32
// ---------------------------------------------------------------------
__global__ void __launch_bounds__(32) conv_kernel(
    const unsigned short* __restrict__ catN, const unsigned short* __restrict__ cwT,
    const float* __restrict__ conv_b, float* __restrict__ hh,
    float* __restrict__ gsum, float* __restrict__ gsumsq)
{
  __shared__ __align__(16) float ct[16*16];   // D-tile transpose staging
  int l = threadIdx.x; int nl = l & 15, hl = l >> 4;
  int w0  = blockIdx.x * 16;
  int h   = blockIdx.y;
  int b   = blockIdx.z >> 3;
  int oc0 = (blockIdx.z & 7) * 16;
  v8f acc;
#pragma unroll
  for (int r = 0; r < 8; ++r) acc[r] = 0.f;

  for (int dy = -1; dy <= 1; ++dy){
    int hs = h + dy;
    if (hs < 0 || hs >= H_) continue;
    const unsigned short* crow = catN + (size_t)(b*H_ + hs)*W_*(2*C_);
    for (int dx = -1; dx <= 1; ++dx){
      int wsrc = w0 + nl + dx;                  // A-row pixel = lane&15
      bool inw = (wsrc >= 0) && (wsrc < W_);
      const unsigned short* apix = crow + (size_t)wsrc*(2*C_);
      const unsigned short* wrow = cwT + (size_t)(((dy+1)*3 + (dx+1))*C_ + oc0 + nl)*(2*C_);
      for (int kc = 0; kc < 2*C_/32; ++kc){
        v16bf a  = inw ? load_frag(apix + kc*32, hl) : zero_frag();
        v16bf bb = load_frag(wrow + kc*32, hl);
        acc = wmma_bf16(a, bb, acc);
      }
    }
  }
  int oc = oc0 + nl;
  float bias = conv_b[oc];
  float s = 0.f, s2 = 0.f;
  // stage in LDS [oc][w] (already that layout: lane holds col=oc, rows=w),
  // then vectorized b128 stores of w-contiguous runs
#pragma unroll
  for (int r = 0; r < 8; ++r){
    float v = acc[r] + bias;
    ct[nl*16 + r + 8*hl] = v;
    s += v; s2 += v*v;
  }
  __syncthreads();
  {
    float* hrow = hh + ((size_t)(b*C_ + oc)*H_ + h)*W_ + w0;
    v4f lo = *(const v4f*)&ct[nl*16 + 8*hl];
    v4f hi = *(const v4f*)&ct[nl*16 + 8*hl + 4];
    *(v4f*)&hrow[8*hl]     = lo;
    *(v4f*)&hrow[8*hl + 4] = hi;
  }
  atomicAdd(&gsum[b*32 + (oc >> 2)], s);
  atomicAdd(&gsumsq[b*32 + (oc >> 2)], s2);
}

// ---------------------------------------------------------------------
// 5) GN stats finalize + normalize/affine/SiLU (v4f vectorized)
// ---------------------------------------------------------------------
__global__ void stats_kernel(const float* __restrict__ gsum, const float* __restrict__ gsumsq,
                             float* __restrict__ mean, float* __restrict__ ginv){
  int i = threadIdx.x;  // 64 = b*32 groups
  const float N = (float)(4 * H_ * W_);
  float m = gsum[i] / N;
  float var = gsumsq[i] / N - m*m;
  mean[i] = m;
  ginv[i] = rsqrtf(var + 1e-6f);
}

__global__ void norm_silu_kernel(const float* __restrict__ hh, const float* __restrict__ mean,
                                 const float* __restrict__ ginv, const float* __restrict__ gamma,
                                 const float* __restrict__ beta, float* __restrict__ out)
{
  const size_t n4 = ((size_t)B_ * C_ * H_ * W_) / 4;
  for (size_t i4 = (size_t)blockIdx.x * blockDim.x + threadIdx.x; i4 < n4;
       i4 += (size_t)gridDim.x * blockDim.x){
    int b  = (int)(i4 >> 22);
    int ch = (int)((i4 >> 15) & 127);
    int gi = b*32 + (ch >> 2);
    float mu = mean[gi], iv = ginv[gi];
    float ga = gamma[ch], be = beta[ch];
    v4f v = *(const v4f*)&hh[i4*4];
    v4f o;
#pragma unroll
    for (int j = 0; j < 4; ++j){
      float y = (v[j] - mu) * iv * ga + be;
      o[j] = y / (1.f + __expf(-y));   // SiLU
    }
    *(v4f*)&out[i4*4] = o;
  }
}

// ---------------------------------------------------------------------
// launcher
// ---------------------------------------------------------------------
extern "C" void kernel_launch(void* const* d_in, const int* in_sizes, int n_in,
                              void* d_out, int out_size, void* d_ws, size_t ws_size,
                              hipStream_t stream) {
  (void)in_sizes; (void)n_in; (void)out_size; (void)ws_size;
  const float* x      = (const float*)d_in[0];
  const float* v      = (const float*)d_in[1];
  const float* Wq     = (const float*)d_in[2];
  const float* Wk     = (const float*)d_in[3];
  const float* Wv     = (const float*)d_in[4];
  const float* Wo     = (const float*)d_in[5];
  const float* bo     = (const float*)d_in[6];
  const float* mask_w = (const float*)d_in[7];
  const float* conv_w = (const float*)d_in[8];
  const float* conv_b = (const float*)d_in[9];
  const float* gamma  = (const float*)d_in[10];
  const float* beta   = (const float*)d_in[11];

  // workspace layout (sizes in bytes, all 16B-aligned)
  char* ws = (char*)d_ws;
  size_t off = 0;
  float* binary = (float*)(ws + off); off += 4096;
  float* gsum   = (float*)(ws + off); off += 256;
  float* gsumsq = (float*)(ws + off); off += 256;
  float* meanp  = (float*)(ws + off); off += 256;
  float* ginvp  = (float*)(ws + off); off += 256 + 3072;            // pad to 8192
  unsigned short* wqT = (unsigned short*)(ws + off); off += (size_t)DA_*DA_*2;     // 128 KB
  unsigned short* woT = (unsigned short*)(ws + off); off += (size_t)H_*DA_*2;      // 128 KB
  unsigned short* wkT = (unsigned short*)(ws + off); off += (size_t)DA_*DC_*2;     // 256 KB
  unsigned short* wvT = (unsigned short*)(ws + off); off += (size_t)DA_*DC_*2;     // 256 KB
  unsigned short* vb  = (unsigned short*)(ws + off); off += (size_t)B_*NC_*DC_*2;  // 1 MB
  unsigned short* cwT = (unsigned short*)(ws + off); off += (size_t)9*C_*2*C_*2;   // 576 KB
  unsigned short* Kp  = (unsigned short*)(ws + off); off += (size_t)B_*NC_*DA_*2;  // 512 KB
  unsigned short* VpT = (unsigned short*)(ws + off); off += (size_t)B_*DA_*NC_*2;  // 512 KB
  unsigned short* attnf = (unsigned short*)(ws + off); off += (size_t)B_*C_*H_*W_*2;   // 64 MB
  unsigned short* catN  = (unsigned short*)(ws + off); off += (size_t)B_*H_*W_*2*C_*2; // 128 MB
  float* hhbuf = (float*)(ws + off);                                             // 128 MB

  float* out_main = (float*)d_out;                     // [2,128,256,512]
  float* out_mask = out_main + (size_t)B_*C_*H_*W_;    // [2,1,512,256]

  // prep
  zero_stats_kernel<<<1, 64, 0, stream>>>(gsum, gsumsq);
  mask_kernel<<<1, H_, 0, stream>>>(mask_w, binary);
  mask_out_kernel<<<(B_*W_*H_/4)/256, 256, 0, stream>>>(binary, out_mask);
  cvt_kernel<<<512, 256, 0, stream>>>(v, vb, B_*NC_*DC_);
  transpose_cvt_kernel<<<(DA_*DA_+255)/256, 256, 0, stream>>>(Wq, wqT, DA_, DA_);
  transpose_cvt_kernel<<<(DA_*H_ +255)/256, 256, 0, stream>>>(Wo, woT, DA_, H_);   // dst[n*256+k]
  transpose_cvt_kernel<<<(DC_*DA_+255)/256, 256, 0, stream>>>(Wk, wkT, DC_, DA_);
  transpose_cvt_kernel<<<(DC_*DA_+255)/256, 256, 0, stream>>>(Wv, wvT, DC_, DA_);
  convw_prep_kernel<<<(9*C_*2*C_+255)/256, 256, 0, stream>>>(conv_w, cwT);

  // compute
  kv_proj_kernel<<<dim3((B_*NC_)/16, DA_/16, 2), 32, 0, stream>>>(vb, wkT, wvT, Kp, VpT);
  attn_kernel<<<dim3(W_/16, C_, B_), 32, 0, stream>>>(x, Kp, VpT, wqT, woT, bo, attnf);
  repack_kernel<<<dim3(W_/64, H_, B_), 256, 0, stream>>>(x, attnf, binary, catN);
  conv_kernel<<<dim3(W_/16, H_, B_*(C_/16)), 32, 0, stream>>>(catN, cwT, conv_b,
                                                              hhbuf, gsum, gsumsq);
  stats_kernel<<<1, 64, 0, stream>>>(gsum, gsumsq, meanp, ginvp);
  norm_silu_kernel<<<8192, 256, 0, stream>>>(hhbuf, meanp, ginvp, gamma, beta, out_main);
}
